// SelfAttention_14328010899852
// MI455X (gfx1250) — compile-verified, matrix operand order checked by reading, not run on hardware
//
#include <hip/hip_runtime.h>

// ---------------------------------------------------------------------------
// MI455X (gfx1250) self-attention: bf16 WMMA + TDM tile staging + DPP softmax.
// B=4, S=2048, E=1024, H=16, D=64.
// ---------------------------------------------------------------------------

#define S_LEN   2048
#define E_DIM   1024
#define NHEADS  16
#define HEAD_D  64

typedef __attribute__((ext_vector_type(16))) __bf16       v16bf;
typedef __attribute__((ext_vector_type(8)))  float        v8f;
typedef __attribute__((ext_vector_type(4)))  unsigned int u32x4;
typedef __attribute__((ext_vector_type(8)))  int          i32x8;
typedef __attribute__((ext_vector_type(4)))  int          i32x4;

union BfPack { v16bf v; u32x4 q[2]; unsigned short s[16]; };

// round-to-nearest-even f32 -> bf16 bits
__device__ __forceinline__ unsigned short f2bf(float f) {
  union { float f; unsigned int u; } x; x.f = f;
  unsigned int u = x.u;
  u += 0x7FFFu + ((u >> 16) & 1u);
  return (unsigned short)(u >> 16);
}
// two floats -> packed bf16 pair (lo, hi)
__device__ __forceinline__ unsigned int f2bf2(float lo, float hi) {
  return (unsigned)f2bf(lo) | ((unsigned)f2bf(hi) << 16);
}

// Build one 16x32 bf16 WMMA operand in the ISA's 16-bit A layout.
// (Also the B layout when the source is stored transposed, i.e. [n][k].)
// Per lane: m = lane&15, half = lane>>4, K-chunks {h*8..+7} and {h*8+16..+23}
// -> two contiguous 16-byte loads.
__device__ __forceinline__ v16bf load_tile16x32(const unsigned short* base, int stride) {
  const int lane = threadIdx.x & 31;
  const int m  = lane & 15;
  const int hf = lane >> 4;
  const unsigned short* p = base + m * stride + hf * 8;
  BfPack u;
  u.q[0] = *(const u32x4*)(p);
  u.q[1] = *(const u32x4*)(p + 16);
  return u.v;
}

__device__ __forceinline__ v8f wmma_bf16(v16bf a, v16bf b, v8f c) {
  return __builtin_amdgcn_wmma_f32_16x16x32_bf16(false, a, false, b, (short)0, c,
                                                 false, false);
}

// ---- DPP row reductions over 16 lanes (a DPP "row"): row_ror 1,2,4,8 -------
__device__ __forceinline__ float row16_max(float x) {
  x = fmaxf(x, __int_as_float(__builtin_amdgcn_update_dpp(
                   0, __float_as_int(x), 0x121, 0xF, 0xF, true)));
  x = fmaxf(x, __int_as_float(__builtin_amdgcn_update_dpp(
                   0, __float_as_int(x), 0x122, 0xF, 0xF, true)));
  x = fmaxf(x, __int_as_float(__builtin_amdgcn_update_dpp(
                   0, __float_as_int(x), 0x124, 0xF, 0xF, true)));
  x = fmaxf(x, __int_as_float(__builtin_amdgcn_update_dpp(
                   0, __float_as_int(x), 0x128, 0xF, 0xF, true)));
  return x;
}
__device__ __forceinline__ float row16_sum(float x) {
  x += __int_as_float(__builtin_amdgcn_update_dpp(0, __float_as_int(x), 0x121, 0xF, 0xF, true));
  x += __int_as_float(__builtin_amdgcn_update_dpp(0, __float_as_int(x), 0x122, 0xF, 0xF, true));
  x += __int_as_float(__builtin_amdgcn_update_dpp(0, __float_as_int(x), 0x124, 0xF, 0xF, true));
  x += __int_as_float(__builtin_amdgcn_update_dpp(0, __float_as_int(x), 0x128, 0xF, 0xF, true));
  return x;
}

// ---- Tensor Data Mover: 2D bf16 tile (tile_d0 x tile_d1 rows) -> LDS -------
// D# per CDNA5 ISA: group0 = {count|flags, lds_addr,
// global_addr[31:0], global_addr[56:32]|type=2}; group1 carries data_size=2B,
// tensor dims (== tile dims: never OOB), tile dims, dim0 stride (elements).
// This toolchain exposes the 6-arg builtin:
//   (u32x4 g0, i32x8 g1, i32x4 g2, i32x4 g3, i32x8 extra, i32 cpol)
#if defined(__has_builtin)
#if __has_builtin(__builtin_amdgcn_tensor_load_to_lds) && \
    __has_builtin(__builtin_amdgcn_s_wait_tensorcnt)
#define USE_TDM 1
#endif
#endif
#ifndef USE_TDM
#define USE_TDM 0
#endif

#if USE_TDM
__device__ __forceinline__ void tdm_load_tile_bf16(void* lds_dst, const void* gsrc,
                                                   unsigned tile_d0, unsigned tile_d1,
                                                   unsigned stride0) {
  const unsigned long long ga = (unsigned long long)gsrc;
  const unsigned lds = (unsigned)(unsigned long long)lds_dst;  // LDS = addr[31:0]
  u32x4 g0;
  g0.x = 1u;                                                  // count=1 (valid)
  g0.y = lds;                                                 // lds_addr
  g0.z = (unsigned)ga;                                        // global_addr[31:0]
  g0.w = (unsigned)((ga >> 32) & 0x01FFFFFFull) | (2u << 30); // [56:32] | type=2
  i32x8 g1 = {};
  g1[0] = (int)(1u << 16);                                    // data_size = 2B
  g1[1] = (int)((tile_d0 & 0xFFFFu) << 16);                   // tensor_dim0 lo
  g1[2] = (int)(((tile_d0 >> 16) & 0xFFFFu) |                 // tensor_dim0 hi
                ((tile_d1 & 0xFFFFu) << 16));                 // tensor_dim1 lo
  g1[3] = (int)(((tile_d1 >> 16) & 0xFFFFu) |                 // tensor_dim1 hi
                ((tile_d0 & 0xFFFFu) << 16));                 // tile_dim0
  g1[4] = (int)(tile_d1 & 0xFFFFu);                           // tile_dim1 (dim2=0)
  g1[5] = (int)stride0;                                       // tensor_dim0_stride
  i32x4 z4 = {};
  i32x8 z8 = {};
  __builtin_amdgcn_tensor_load_to_lds(g0, g1, z4, z4, z8, 0);
}
#endif

// ---------------------------------------------------------------------------
// GEMM: C[M=8192, N=1024] = A[M,1024] x W[1024,1024] + bias
// 256 threads (8 waves), tile 128x128, K-step 32; wave = 2x4 WMMA accum tiles.
// A_IS_F32: A read as fp32 and converted while staging; else A is bf16 bits
// and the whole 128x32 A tile is staged by one TDM tensor_load_to_lds.
// OUT_QKV : write bf16 into [B,H,S,D]; else fp32 row-major [M,N].
// ---------------------------------------------------------------------------
template <int A_IS_F32, int OUT_QKV>
__global__ __launch_bounds__(256) void gemm_bf16_wmma(
    const void* __restrict__ A_, const float* __restrict__ W,
    const float* __restrict__ bias, void* __restrict__ out_) {
  __shared__ unsigned short As[128][32];  // [m][k]
  __shared__ unsigned short Wt[128][32];  // [n][k] (transposed weight tile)

  const int tid  = threadIdx.x;
  const int lane = tid & 31;
  const int wave = tid >> 5;
  const int hf   = lane >> 4;
  const int bm   = blockIdx.x * 128;
  const int bn   = blockIdx.y * 128;
  const int wm   = (wave & 3) * 32;
  const int wn   = (wave >> 2) * 64;

  v8f acc[2][4] = {};

  for (int k0 = 0; k0 < E_DIM; k0 += 32) {
    __syncthreads();
    bool tdm_issued = false;
    if constexpr (!A_IS_F32) {
#if USE_TDM
      if (tid < 32) {  // one TDM issue per block (EXEC ignored; wave-level op)
        tdm_load_tile_bf16(&As[0][0],
                           (const unsigned short*)A_ + (size_t)bm * E_DIM + k0,
                           /*tile_d0=*/32, /*tile_d1=*/128, /*stride0=*/E_DIM);
      }
      tdm_issued = true;
#endif
    }
    if constexpr (A_IS_F32) {  // stage A tile (128x32) with f32->bf16 convert
      const int row  = tid >> 1;
      const int koff = (tid & 1) * 16;
      const float* src = (const float*)A_ + (size_t)(bm + row) * E_DIM + k0 + koff;
      if (k0 + 32 < E_DIM) __builtin_prefetch(src + 32, 0, 1);
      unsigned int* dst = (unsigned int*)&As[row][koff];
      #pragma unroll
      for (int j = 0; j < 8; ++j) dst[j] = f2bf2(src[2 * j], src[2 * j + 1]);
    } else if (!tdm_issued) {  // fallback if TDM builtin unavailable
      const int row  = tid >> 1;
      const int koff = (tid & 1) * 16;
      const unsigned short* src =
          (const unsigned short*)A_ + (size_t)(bm + row) * E_DIM + k0 + koff;
      *(u32x4*)&As[row][koff]     = *(const u32x4*)src;
      *(u32x4*)&As[row][koff + 8] = *(const u32x4*)(src + 8);
    }
    {  // stage W tile transposed (32k x 128n -> Wt[n][k]), coalesced in n
      const int k    = tid >> 3;
      const int noff = (tid & 7) * 16;
      const float* src = W + (size_t)(k0 + k) * E_DIM + bn + noff;
      if (k0 + 32 < E_DIM) __builtin_prefetch(src + 32 * E_DIM, 0, 1);
      #pragma unroll
      for (int j = 0; j < 16; ++j) Wt[noff + j][k] = f2bf(src[j]);
    }
#if USE_TDM
    if (!A_IS_F32 && tid < 32) __builtin_amdgcn_s_wait_tensorcnt(0);
#endif
    __syncthreads();

    v16bf a[2], b[4];
    #pragma unroll
    for (int mt = 0; mt < 2; ++mt) a[mt] = load_tile16x32(&As[wm + mt * 16][0], 32);
    #pragma unroll
    for (int nt = 0; nt < 4; ++nt) b[nt] = load_tile16x32(&Wt[wn + nt * 16][0], 32);
    #pragma unroll
    for (int mt = 0; mt < 2; ++mt)
      #pragma unroll
      for (int nt = 0; nt < 4; ++nt)
        acc[mt][nt] = wmma_bf16(a[mt], b[nt], acc[mt][nt]);
  }

  // Epilogue. C layout: VGPR r -> row r + 8*half, col = lane&15 within tile.
  #pragma unroll
  for (int mt = 0; mt < 2; ++mt)
    #pragma unroll
    for (int nt = 0; nt < 4; ++nt)
      #pragma unroll
      for (int r = 0; r < 8; ++r) {
        const int row = bm + wm + mt * 16 + r + 8 * hf;  // global M index (b*S+s)
        const int col = bn + wn + nt * 16 + (lane & 15); // global N index (h*64+d)
        const float v = acc[mt][nt][r] + bias[col];
        if constexpr (OUT_QKV) {
          const int b_ = row >> 11, s = row & (S_LEN - 1);
          const int h = col >> 6, d = col & (HEAD_D - 1);
          ((unsigned short*)out_)[(((size_t)(b_ * NHEADS + h) * S_LEN + s) << 6) + d] =
              f2bf(v);
        } else {
          ((float*)out_)[(size_t)row * E_DIM + col] = v;
        }
      }
}

// ---------------------------------------------------------------------------
// Flash attention. Grid: (S/64 query-blocks, B*H). Block = 128 threads = 4 waves.
// Wave w owns query rows qblock*64 + w*16 .. +15, accumulates O[16,64] in f32.
// K tile staged by TDM; V tile staged transposed by threads; QK^T (2 WMMA
// k-steps) -> online softmax (DPP row reductions) -> P through per-wave LDS
// to A-layout -> PV (4 WMMAs, K=32).
// ---------------------------------------------------------------------------
__global__ __launch_bounds__(128) void flash_attn_wmma(
    const unsigned short* __restrict__ Q, const unsigned short* __restrict__ Kb,
    const unsigned short* __restrict__ Vb, unsigned short* __restrict__ Y) {
  __shared__ unsigned short lK[32][64];     // [key_local][d]
  __shared__ unsigned short lV[64][32];     // [d][key_local] (transposed)
  __shared__ unsigned short lP[4][16][32];  // per-wave P tile [qrow][key_local]

  const int tid    = threadIdx.x;
  const int lane   = tid & 31;
  const int wave   = tid >> 5;
  const int hf     = lane >> 4;
  const int qblock = blockIdx.x;  // 0..31
  const int bh     = blockIdx.y;  // 0..63
  const int qbase  = qblock * 64 + wave * 16;

  // Q operand for this wave's 16 rows (16x64 bf16), loaded once.
  const unsigned short* qrow = Q + ((size_t)bh * S_LEN + qbase) * HEAD_D;
  v16bf aq[2];
  aq[0] = load_tile16x32(qrow, HEAD_D);       // d 0..31
  aq[1] = load_tile16x32(qrow + 32, HEAD_D);  // d 32..63

  v8f acc[4] = {};
  float run_max[8], run_sum[8];
  #pragma unroll
  for (int r = 0; r < 8; ++r) { run_max[r] = -1e30f; run_sum[r] = 0.0f; }

  const float sc = 0.125f;         // 1/sqrt(64)
  const int nkb = qblock * 2 + 2;  // key blocks covering keys <= qblock*64+63

  for (int kb = 0; kb < nkb; ++kb) {
    __syncthreads();
    const size_t blk0 = ((size_t)bh * S_LEN + kb * 32) * HEAD_D;
#if USE_TDM
    if (tid < 32) {  // whole 32x64 bf16 K tile in one TDM op
      tdm_load_tile_bf16(&lK[0][0], Kb + blk0, /*d0=*/HEAD_D, /*d1=*/32, HEAD_D);
    }
#endif
    {  // cooperative V load, transposed into lV[d][key]
      const int key = tid >> 2;
      const int d0  = (tid & 3) * 16;
      const unsigned short* vs = Vb + blk0 + (size_t)key * HEAD_D + d0;
      BfPack t;
      t.q[0] = *(const u32x4*)vs;
      t.q[1] = *(const u32x4*)(vs + 8);
      #pragma unroll
      for (int j = 0; j < 16; ++j) lV[d0 + j][key] = t.s[j];
#if !USE_TDM
      const unsigned short* ks = Kb + blk0 + (size_t)key * HEAD_D + d0;
      *(u32x4*)&lK[key][d0]     = *(const u32x4*)ks;
      *(u32x4*)&lK[key][d0 + 8] = *(const u32x4*)(ks + 8);
#endif
    }
#if USE_TDM
    if (tid < 32) __builtin_amdgcn_s_wait_tensorcnt(0);
#endif
    __syncthreads();

    // S[16x32] = Q * K^T : two 16x16 column tiles, each 2 k-steps over d.
    v8f s0 = {}, s1 = {};
    {
      v16bf b00 = load_tile16x32(&lK[0][0], 64);
      v16bf b01 = load_tile16x32(&lK[0][32], 64);
      s0 = wmma_bf16(aq[0], b00, s0);
      s0 = wmma_bf16(aq[1], b01, s0);
      v16bf b10 = load_tile16x32(&lK[16][0], 64);
      v16bf b11 = load_tile16x32(&lK[16][32], 64);
      s1 = wmma_bf16(aq[0], b10, s1);
      s1 = wmma_bf16(aq[1], b11, s1);
    }

    // Online softmax per row (row r+8*hf lives across the 16 lanes of a half).
    #pragma unroll
    for (int r = 0; r < 8; ++r) {
      const int qi = qbase + r + 8 * hf;
      const int c0 = kb * 32 + (lane & 15);
      const int c1 = c0 + 16;
      float x0 = (c0 <= qi) ? s0[r] * sc : -1e30f;
      float x1 = (c1 <= qi) ? s1[r] * sc : -1e30f;
      const float cur = row16_max(fmaxf(x0, x1));
      const float nm = fmaxf(run_max[r], cur);
      const float fs = __expf(run_max[r] - nm);
      const float p0 = __expf(x0 - nm);
      const float p1 = __expf(x1 - nm);
      const float ps = row16_sum(p0 + p1);
      run_sum[r] = run_sum[r] * fs + ps;
      run_max[r] = nm;
      #pragma unroll
      for (int nt = 0; nt < 4; ++nt) acc[nt][r] *= fs;
      lP[wave][r + 8 * hf][lane & 15]        = f2bf(p0);
      lP[wave][r + 8 * hf][16 + (lane & 15)] = f2bf(p1);
    }

    // O[16x64] += P[16x32] * V[32x64] (K=32, one WMMA per d-tile).
    // Intra-wave LDS RAW: DS ops are in-order per wave, no barrier needed.
    v16bf ap = load_tile16x32(&lP[wave][0][0], 32);
    #pragma unroll
    for (int nt = 0; nt < 4; ++nt) {
      v16bf bv = load_tile16x32(&lV[nt * 16][0], 32);
      acc[nt] = wmma_bf16(ap, bv, acc[nt]);
    }
  }

  // Normalize and write y as bf16 into [B,S,E] (E = h*64 + d).
  const int b_ = bh >> 4, h = bh & (NHEADS - 1);
  #pragma unroll
  for (int nt = 0; nt < 4; ++nt)
    #pragma unroll
    for (int r = 0; r < 8; ++r) {
      const int s   = qbase + r + 8 * hf;
      const int col = h * HEAD_D + nt * 16 + (lane & 15);
      const float v = acc[nt][r] / run_sum[r];
      Y[((size_t)(b_ * S_LEN + s)) * E_DIM + col] = f2bf(v);
    }
}

// ---------------------------------------------------------------------------
extern "C" void kernel_launch(void* const* d_in, const int* in_sizes, int n_in,
                              void* d_out, int out_size, void* d_ws, size_t ws_size,
                              hipStream_t stream) {
  (void)in_sizes; (void)n_in; (void)out_size; (void)ws_size;
  const float* key_in   = (const float*)d_in[0];
  const float* value_in = (const float*)d_in[1];
  const float* query_in = (const float*)d_in[2];
  const float* Wk = (const float*)d_in[3];
  const float* bk = (const float*)d_in[4];
  const float* Wq = (const float*)d_in[5];
  const float* bq = (const float*)d_in[6];
  const float* Wv = (const float*)d_in[7];
  const float* bv = (const float*)d_in[8];
  const float* Wp = (const float*)d_in[9];
  const float* bp = (const float*)d_in[10];

  const size_t elems = (size_t)4 * S_LEN * E_DIM;  // 8,388,608 bf16 per buffer
  unsigned short* qb = (unsigned short*)d_ws;
  unsigned short* kb = qb + elems;
  unsigned short* vb = kb + elems;
  unsigned short* yb = vb + elems;

  const dim3 ggrid(64, 8), gblock(256);
  gemm_bf16_wmma<1, 1><<<ggrid, gblock, 0, stream>>>((const void*)query_in, Wq, bq, (void*)qb);
  gemm_bf16_wmma<1, 1><<<ggrid, gblock, 0, stream>>>((const void*)key_in,   Wk, bk, (void*)kb);
  gemm_bf16_wmma<1, 1><<<ggrid, gblock, 0, stream>>>((const void*)value_in, Wv, bv, (void*)vb);

  const dim3 agrid(S_LEN / 64, 4 * NHEADS), ablock(128);
  flash_attn_wmma<<<agrid, ablock, 0, stream>>>(qb, kb, vb, yb);

  gemm_bf16_wmma<0, 0><<<ggrid, gblock, 0, stream>>>((const void*)yb, Wp, bp, d_out);
}
